// Network_28544352649223
// MI455X (gfx1250) — matrix operand order repaired and checked by reading, not compiled, not run on hardware
//
#include <hip/hip_runtime.h>
#include <hip/hip_fp16.h>

typedef __attribute__((ext_vector_type(16))) _Float16 v16h;
typedef __attribute__((ext_vector_type(8)))  _Float16 v8h;
typedef __attribute__((ext_vector_type(4)))  _Float16 v4h;
typedef __attribute__((ext_vector_type(2)))  _Float16 v2h;
typedef __attribute__((ext_vector_type(8)))  float    v8f;
typedef __attribute__((ext_vector_type(4)))  int      v4i;

// address-space-qualified int4 for the async global->LDS builtin
typedef __attribute__((address_space(1))) v4i gv4i;
typedef __attribute__((address_space(3))) v4i lv4i;

#define WMMA_F16(a, b, c) \
  __builtin_amdgcn_wmma_f32_16x16x32_f16(false, (a), false, (b), (short)0, (c), false, false)

// ---- CDNA5 async global->LDS copy (ASYNCcnt path), with safe fallback -------

#if defined(__gfx1250__) && __has_builtin(__builtin_amdgcn_global_load_async_to_lds_b128)
#define HAVE_ASYNC_LDS 1
#else
#define HAVE_ASYNC_LDS 0
#endif

static __device__ __forceinline__ void async_copy16(_Float16* lds, const _Float16* g) {
#if HAVE_ASYNC_LDS
  __builtin_amdgcn_global_load_async_to_lds_b128((gv4i*)(g), (lv4i*)(lds), 0, 0);
#else
  *(v8h*)lds = *(const v8h*)g;
#endif
}

static __device__ __forceinline__ void wait_async() {
#if HAVE_ASYNC_LDS
#if __has_builtin(__builtin_amdgcn_s_wait_asynccnt)
  __builtin_amdgcn_s_wait_asynccnt(0);
#else
  asm volatile("s_wait_asynccnt 0" ::: "memory");
#endif
#endif
}

// ---- WMMA fragment loaders (ISA 7.12.2 layouts, wave32) ----------------------
// All loaders read contiguous 16B chunks per lane -> b128 loads.

// A fragment 16x32 from row-major M x K source (ld multiple of 8).
static __device__ __forceinline__ v16h load_frag_rowmajor(const _Float16* p, int ld) {
  const int lane = threadIdx.x & 31;
  const int m = lane & 15;
  const int kb = (lane < 16) ? 0 : 8;
  const _Float16* r = p + m * ld + kb;
  v8h lo = *(const v8h*)(r);
  v8h hi = *(const v8h*)(r + 16);
  return __builtin_shufflevector(lo, hi, 0, 1, 2, 3, 4, 5, 6, 7,
                                 8, 9, 10, 11, 12, 13, 14, 15);
}

// B fragment 32x16 from N-major source (row = n, ld K-elems per row).
static __device__ __forceinline__ v16h load_frag_nmajor(const _Float16* p, int ld) {
  const int lane = threadIdx.x & 31;
  const int n = lane & 15;
  const int kb = (lane < 16) ? 0 : 8;
  const _Float16* r = p + n * ld + kb;
  v8h lo = *(const v8h*)(r);
  v8h hi = *(const v8h*)(r + 16);
  return __builtin_shufflevector(lo, hi, 0, 1, 2, 3, 4, 5, 6, 7,
                                 8, 9, 10, 11, 12, 13, 14, 15);
}

static __device__ __forceinline__ v4h cvt4(const float4 f) {
  v4h h;
  h[0] = (_Float16)f.x; h[1] = (_Float16)f.y;
  h[2] = (_Float16)f.z; h[3] = (_Float16)f.w;
  return h;
}

// ---- Kernel 1: fused QKV projection  qkv[b,h,t,d] = sum_c x[b,t,c]*w[h,d,c] --
// Block tile 64(M) x 192(N = all of Q|K|V); 8 waves as 4x2; wave = 16x96.
// x tile staged once and shared by all three outputs.

__global__ __launch_bounds__(256) void qkv_kernel(
    const float* __restrict__ x, const float* __restrict__ w_qkv,
    _Float16* __restrict__ q_out, _Float16* __restrict__ k_out,
    _Float16* __restrict__ v_out) {
  int id = blockIdx.x;
  const int tb = id % 32; id /= 32;    // T/64 row tiles
  const int bh = id;                   // b*16 + h
  const int h = bh & 15;

  const int tid = threadIdx.x;
  const int wave = tid >> 5;
  const int mw = wave >> 1, nw = wave & 1;   // 4(M) x 2(N=96 each)
  const int lane = tid & 31;
  const int nlane = lane & 15;
  const int mbase = (lane < 16) ? 0 : 8;

  __shared__ __align__(16) _Float16 alds[64 * 64];    // A: row-major 64x64
  __shared__ __align__(16) _Float16 blds[192 * 64];   // B: N-major 192n x 64k

  const float* xr = x + ((size_t)(bh >> 4) * 2048 + (size_t)tb * 64) * 1024;
  const float* wr = w_qkv + (size_t)h * 192 * 1024;

  v8f acc[6] = {};
  for (int kc = 0; kc < 1024; kc += 64) {
    __syncthreads();
#pragma unroll
    for (int t = 0; t < 4; ++t) {                 // A: 64x64 f32->f16
      int idx = tid + t * 256;
      int r = idx >> 4, c4 = (idx & 15) << 2;
      float4 f = *(const float4*)(xr + (size_t)r * 1024 + kc + c4);
      *(v4h*)(alds + r * 64 + c4) = cvt4(f);
    }
#pragma unroll
    for (int t = 0; t < 12; ++t) {                // B: 192x64 f32->f16
      int idx = tid + t * 256;
      int r = idx >> 4, c4 = (idx & 15) << 2;
      float4 f = *(const float4*)(wr + (size_t)r * 1024 + kc + c4);
      *(v4h*)(blds + r * 64 + c4) = cvt4(f);
    }
    __syncthreads();
    v16h a0 = load_frag_rowmajor(alds + mw * 16 * 64, 64);
    v16h a1 = load_frag_rowmajor(alds + mw * 16 * 64 + 32, 64);
#pragma unroll
    for (int ns = 0; ns < 6; ++ns) {
      const _Float16* bp = blds + (nw * 96 + ns * 16) * 64;
      v16h b0 = load_frag_nmajor(bp, 64);
      v16h b1 = load_frag_nmajor(bp + 32, 64);
      acc[ns] = WMMA_F16(a0, b0, acc[ns]);
      acc[ns] = WMMA_F16(a1, b1, acc[ns]);
    }
  }

  const int t0 = tb * 64 + mw * 16 + mbase;
#pragma unroll
  for (int ns = 0; ns < 6; ++ns) {
    const int d = nw * 96 + ns * 16 + nlane;   // 0..191; `which` uniform per ns
    const int which = d >> 6;
    const int hs = d & 63;
    _Float16* dst = (which == 0) ? q_out : (which == 1) ? k_out : v_out;
    dst += (size_t)bh * 2048 * 64 + hs;
#pragma unroll
    for (int i = 0; i < 8; ++i)
      dst[(size_t)(t0 + i) * 64] = (_Float16)acc[ns][i];
  }
}

// ---- Kernel 2: causal flash attention, per (b,h) ----------------------------

__global__ __launch_bounds__(256) void attn_kernel(
    const _Float16* __restrict__ qbuf, const _Float16* __restrict__ kbuf,
    const _Float16* __restrict__ vbuf, _Float16* __restrict__ att) {
  const int qblk = blockIdx.x;   // 0..15, 128 query rows each
  const int bh = blockIdx.y;     // 0..31
  const int h = bh & 15;
  const int b = bh >> 4;

  const int tid = threadIdx.x;
  const int wave = tid >> 5;     // each wave owns 16 query rows
  const int lane = tid & 31;
  const int n = lane & 15;
  const int mbase = (lane < 16) ? 0 : 8;

  __shared__ __align__(16) _Float16 klds[64 * 64];    // [key][d] row-major
  __shared__ __align__(16) _Float16 vtlds[64 * 64];   // transposed: [d][key]
  __shared__ __align__(16) _Float16 plds[8 * 16 * 64];

  const int qs = qblk * 128 + wave * 16;
  const float SCL = 0.125f * 1.44269504088896f;   // 1/sqrt(HS) * log2(e)

  const _Float16* qg = qbuf + ((size_t)bh * 2048 + qs) * 64;
  v16h aQ0 = load_frag_rowmajor(qg, 64);        // head-dim K 0..31
  v16h aQ1 = load_frag_rowmajor(qg + 32, 64);   // 32..63

  v8f o[4] = {};
  float mrow[8], lrow[8];
#pragma unroll
  for (int i = 0; i < 8; ++i) { mrow[i] = -3.0e38f; lrow[i] = 0.f; }

  const int nkt = 2 * qblk + 2;          // key tiles up to & incl. diagonal
  for (int kt = 0; kt < nkt; ++kt) {
    const int kb0 = kt * 64;
    __syncthreads();
    const _Float16* kg = kbuf + ((size_t)bh * 2048 + kb0) * 64;
    const _Float16* vg = vbuf + ((size_t)bh * 2048 + kb0) * 64;
    // K tile: pure 16B copies -> async global->LDS (ASYNCcnt), overlapped
    async_copy16(klds + tid * 8, kg + tid * 8);
    async_copy16(klds + 2048 + tid * 8, kg + 2048 + tid * 8);
    {
      // V: read two keys x 8 dims, write transposed as packed b32 pairs
      const int k2 = (tid >> 3) << 1;        // even key 0..62
      const int d8 = (tid & 7) << 3;
      v8h va = *(const v8h*)(vg + k2 * 64 + d8);
      v8h vb = *(const v8h*)(vg + (k2 + 1) * 64 + d8);
#pragma unroll
      for (int j = 0; j < 8; ++j) {
        v2h pr; pr[0] = va[j]; pr[1] = vb[j];
        *(v2h*)(vtlds + (d8 + j) * 64 + k2) = pr;
      }
    }
    wait_async();
    __syncthreads();

    // S = Q K^T  (16 q-rows x 64 keys per wave); K tile is key(n)-major
    v8f s[4];
#pragma unroll
    for (int ntile = 0; ntile < 4; ++ntile) {
      v8f z = {};
      v16h b0 = load_frag_nmajor(klds + ntile * 16 * 64, 64);
      v16h b1 = load_frag_nmajor(klds + ntile * 16 * 64 + 32, 64);
      z = WMMA_F16(aQ0, b0, z);
      z = WMMA_F16(aQ1, b1, z);
      s[ntile] = z;
    }

    // scale into log2 domain + causal mask
#pragma unroll
    for (int ntile = 0; ntile < 4; ++ntile)
#pragma unroll
      for (int i = 0; i < 8; ++i) {
        float sv = s[ntile][i] * SCL;
        if (kb0 + ntile * 16 + n > qs + mbase + i) sv = -3.0e38f;
        s[ntile][i] = sv;
      }

    // online softmax (base-2): each row lives in one 16-lane half
    float rmax[8], rsum[8];
#pragma unroll
    for (int i = 0; i < 8; ++i)
      rmax[i] = fmaxf(fmaxf(s[0][i], s[1][i]), fmaxf(s[2][i], s[3][i]));
#pragma unroll
    for (int msk = 1; msk <= 8; msk <<= 1)
#pragma unroll
      for (int i = 0; i < 8; ++i)
        rmax[i] = fmaxf(rmax[i], __shfl_xor(rmax[i], msk, 32));

    float corr[8];
#pragma unroll
    for (int i = 0; i < 8; ++i) {
      float mn = fmaxf(mrow[i], rmax[i]);
      corr[i] = exp2f(mrow[i] - mn);
      mrow[i] = mn;
      rsum[i] = 0.f;
    }
#pragma unroll
    for (int ntile = 0; ntile < 4; ++ntile)
#pragma unroll
      for (int i = 0; i < 8; ++i) {
        float p = exp2f(s[ntile][i] - mrow[i]);
        s[ntile][i] = p;
        rsum[i] += p;
      }
#pragma unroll
    for (int msk = 1; msk <= 8; msk <<= 1)
#pragma unroll
      for (int i = 0; i < 8; ++i) rsum[i] += __shfl_xor(rsum[i], msk, 32);
#pragma unroll
    for (int i = 0; i < 8; ++i) lrow[i] = lrow[i] * corr[i] + rsum[i];
#pragma unroll
    for (int ntile = 0; ntile < 4; ++ntile)
#pragma unroll
      for (int i = 0; i < 8; ++i) o[ntile][i] *= corr[i];

    // stage P as f16 in per-wave LDS (C-layout -> row-major for A fragments)
    _Float16* pl = plds + wave * 16 * 64;
#pragma unroll
    for (int ntile = 0; ntile < 4; ++ntile)
#pragma unroll
      for (int i = 0; i < 8; ++i)
        pl[(mbase + i) * 64 + ntile * 16 + n] = (_Float16)s[ntile][i];

    v16h aP0 = load_frag_rowmajor(pl, 64);
    v16h aP1 = load_frag_rowmajor(pl + 32, 64);
#pragma unroll
    for (int ntile = 0; ntile < 4; ++ntile) {
      v16h bv0 = load_frag_nmajor(vtlds + ntile * 16 * 64, 64);
      v16h bv1 = load_frag_nmajor(vtlds + ntile * 16 * 64 + 32, 64);
      o[ntile] = WMMA_F16(aP0, bv0, o[ntile]);
      o[ntile] = WMMA_F16(aP1, bv1, o[ntile]);
    }
  }

  // normalize and store heads-concatenated: att[b*T + t, h*64 + d]
  _Float16* outp = att + ((size_t)b * 2048 + qs) * 1024 + h * 64;
#pragma unroll
  for (int ntile = 0; ntile < 4; ++ntile)
#pragma unroll
    for (int i = 0; i < 8; ++i) {
      float ov = o[ntile][i] / lrow[i];
      outp[(size_t)(mbase + i) * 1024 + ntile * 16 + n] = (_Float16)ov;
    }
}

// ---- Kernel 3: output projection  out = att @ W^T + bias --------------------
// Block tile 64(M) x 128(N); 8 waves as 4x2; wave = 16x64 (4 N-subtiles).

__global__ __launch_bounds__(256) void proj_kernel(
    const _Float16* __restrict__ att, const float* __restrict__ w_proj,
    const float* __restrict__ b_proj, float* __restrict__ out) {
  const int nb = blockIdx.x & 7;    // 8 N-tiles of 128
  const int mb = blockIdx.x >> 3;   // 64 M-tiles of 64
  const int tid = threadIdx.x;
  const int wave = tid >> 5;
  const int mw = wave >> 1, nw = wave & 1;
  const int lane = tid & 31;
  const int n = lane & 15;
  const int mbase = (lane < 16) ? 0 : 8;

  __shared__ __align__(16) _Float16 alds[64 * 64];    // A row-major (f16 copies)
  __shared__ __align__(16) _Float16 blds[128 * 64];   // B N-major (f32->f16)

  const _Float16* ar = att + (size_t)(mb * 64) * 1024;
  const float* wr = w_proj + (size_t)(nb * 128) * 1024;

  v8f acc[4] = {};
  for (int kc = 0; kc < 1024; kc += 64) {
    __syncthreads();
#pragma unroll
    for (int t = 0; t < 2; ++t) {                 // A: async 16B lane copies
      int idx = tid + t * 256;
      int r = idx >> 3, c8 = (idx & 7) << 3;
      async_copy16(alds + r * 64 + c8, ar + (size_t)r * 1024 + kc + c8);
    }
#pragma unroll
    for (int t = 0; t < 8; ++t) {                 // B: 128x64 f32->f16
      int idx = tid + t * 256;
      int r = idx >> 4, c4 = (idx & 15) << 2;
      float4 f = *(const float4*)(wr + (size_t)r * 1024 + kc + c4);
      *(v4h*)(blds + r * 64 + c4) = cvt4(f);
    }
    wait_async();
    __syncthreads();
    v16h a0 = load_frag_rowmajor(alds + mw * 16 * 64, 64);
    v16h a1 = load_frag_rowmajor(alds + mw * 16 * 64 + 32, 64);
#pragma unroll
    for (int ns = 0; ns < 4; ++ns) {
      const _Float16* bp = blds + (nw * 64 + ns * 16) * 64;
      v16h b0 = load_frag_nmajor(bp, 64);
      v16h b1 = load_frag_nmajor(bp + 32, 64);
      acc[ns] = WMMA_F16(a0, b0, acc[ns]);
      acc[ns] = WMMA_F16(a1, b1, acc[ns]);
    }
  }

  const int r0 = mb * 64 + mw * 16 + mbase;
#pragma unroll
  for (int ns = 0; ns < 4; ++ns) {
    const int col = nb * 128 + nw * 64 + ns * 16 + n;
    const float bias = b_proj[col];
#pragma unroll
    for (int i = 0; i < 8; ++i)
      out[(size_t)(r0 + i) * 1024 + col] = acc[ns][i] + bias;
  }
}

// ---- host launch -------------------------------------------------------------

extern "C" void kernel_launch(void* const* d_in, const int* in_sizes, int n_in,
                              void* d_out, int out_size, void* d_ws, size_t ws_size,
                              hipStream_t stream) {
  const float* x      = (const float*)d_in[0];   // [2,2048,1024]
  const float* w_qkv  = (const float*)d_in[1];   // [16,192,1024]
  const float* w_proj = (const float*)d_in[2];   // [1024,1024]
  const float* b_proj = (const float*)d_in[3];   // [1024]
  float* out = (float*)d_out;                    // [2,2048,1024]

  char* ws = (char*)d_ws;
  _Float16* qb  = (_Float16*)(ws);                              // 8 MB
  _Float16* kb  = (_Float16*)(ws + (size_t)8  * 1024 * 1024);   // 8 MB
  _Float16* vb  = (_Float16*)(ws + (size_t)16 * 1024 * 1024);   // 8 MB
  _Float16* att = (_Float16*)(ws + (size_t)24 * 1024 * 1024);   // 8 MB

  // B*H * (T/64) = 2*16*32 = 1024 blocks (Q,K,V fused)
  qkv_kernel<<<dim3(1024), dim3(256), 0, stream>>>(x, w_qkv, qb, kb, vb);
  attn_kernel<<<dim3(16, 32), dim3(256), 0, stream>>>(qb, kb, vb, att);
  // 64 M-tiles x 8 N-tiles of 128
  proj_kernel<<<dim3(512), dim3(256), 0, stream>>>(att, w_proj, b_proj, out);
}
